// DNRI_44452911514389
// MI455X (gfx1250) — matrix operand order repaired and checked by reading, not compiled, not run on hardware
//
#include <hip/hip_runtime.h>

#define B_  32
#define V_  50
#define D_  6
#define H_  256
#define ET_ 4
#define T_  3            // ET-1 (skip_first)
#define E_  (V_*(V_-1))  // 2450
#define MT  64           // padded edge-rows per receiver (49 -> 64)
#define KS  32           // K per WMMA step (bf16 16x16x32)

// d_ws layout (u16 elements):
#define WS_W1   0                         // 3*16 blocks * 8192 = 393216
#define WS_W2   393216                    // 3*8  blocks * 8192 = 196608
#define WS_HID  589824                    // B*V*H = 409600
#define WS_TOT  999424

typedef __attribute__((ext_vector_type(16))) __bf16 v16bf;
typedef __attribute__((ext_vector_type(8)))  float  v8f;
typedef unsigned int  u32;
typedef unsigned short u16;

__device__ __forceinline__ u16 f2bf(float f) {
    u32 u = __builtin_bit_cast(u32, f);
    u += 0x7fffu + ((u >> 16) & 1u);           // round-to-nearest-even
    return (u16)(u >> 16);
}

union Frag { uint4 q[2]; v16bf v; };

__device__ __forceinline__ void wait_async0() {
    asm volatile("s_wait_asynccnt 0x0" ::: "memory");
}
__device__ __forceinline__ u32 lds_off32(const void* p) {
    return (u32)(uintptr_t)p;   // flat shared ptr: low 32 bits = LDS offset
}
// async global->LDS 16B/lane, GVS form: uniform SGPR base + 32-bit lane offset
__device__ __forceinline__ void async_cp16_s(u32 lds_off, u32 voff, const u16* sbase) {
    asm volatile("global_load_async_to_lds_b128 %0, %1, %2"
                 :: "v"(lds_off), "v"(voff), "s"(sbase) : "memory");
}

// =====================================================================
// Prep: convert W1/W2 to bf16 in WMMA-fragment image blocks, hidden->bf16
// Fragment image for a 32x256 K-slice: element (k,n) at
//   ((n/16)*32 + (k/16)*16 + n%16)*16 + k%16      (16 KB per block)
// =====================================================================
__global__ __launch_bounds__(256)
void dnri_prep(const float* __restrict__ W1, const float* __restrict__ W2,
               const float* __restrict__ hidden, u16* __restrict__ ws)
{
    int i = blockIdx.x * 256 + threadIdx.x;
    if (i < 393216) {                       // W1: [3][256][512]
        int t = i >> 17, r = i & 131071;
        int n = r >> 9, ccol = r & 511;
        int kk = ccol >> 5, k = ccol & 31;
        int pos = (((n >> 4) * 32 + ((k >> 4) << 4) + (n & 15)) << 4) + (k & 15);
        ws[WS_W1 + (size_t)(t * 16 + kk) * 8192 + pos] = f2bf(W1[i]);
    } else if (i < 589824) {                // W2: [3][256][256]
        int j = i - 393216;
        int t = j >> 16, r = j & 65535;
        int n = r >> 8, ccol = r & 255;
        int kk = ccol >> 5, k = ccol & 31;
        int pos = (((n >> 4) * 32 + ((k >> 4) << 4) + (n & 15)) << 4) + (k & 15);
        ws[WS_W2 + (size_t)(t * 8 + kk) * 8192 + pos] = f2bf(W2[j]);
    } else if (i < WS_TOT) {                // hidden: [B*V*H]
        int j = i - 589824;
        ws[WS_HID + j] = f2bf(hidden[j]);
    }
}

// =====================================================================
// Main fused kernel: one block per (batch, receiver node)
// =====================================================================
__global__ __launch_bounds__(256)
void dnri_node_kernel(const float* __restrict__ inputs, const float* __restrict__ hidden,
                      const float* __restrict__ edges,
                      const float* __restrict__ b1,  const float* __restrict__ b2,
                      const float* __restrict__ Whr, const float* __restrict__ Whi,
                      const float* __restrict__ Whh,
                      const float* __restrict__ Wir, const float* __restrict__ bir,
                      const float* __restrict__ Wii, const float* __restrict__ bii,
                      const float* __restrict__ Win, const float* __restrict__ bin_w,
                      const float* __restrict__ Wo1, const float* __restrict__ bo1,
                      const float* __restrict__ Wo2, const float* __restrict__ bo2,
                      const float* __restrict__ Wo3, const float* __restrict__ bo3,
                      const u16* __restrict__ wsb,
                      float* __restrict__ pred_out, float* __restrict__ hid_out)
{
    __shared__ u16  sAfull[16][MT * KS];  // all 16 A K-slices, bf16 row-major (64 KB)
    __shared__ u16  sB[2][KS * H_];       // B slices, fragment image, dbl     (32 KB)
    __shared__ u16  sM[MT * H_];          // GEMM1 output, bf16 row-major      (32 KB)
    __shared__ float sAcc[H_];
    __shared__ float sEw[T_][MT];
    __shared__ float sVec[H_];
    __shared__ float sVec2[H_];
    __shared__ float sX[D_];

    const int bv   = blockIdx.x;
    const int b    = bv / V_;
    const int v    = bv % V_;
    const int tid  = threadIdx.x;
    const int lane = tid & 31;
    const int wave = tid >> 5;
    const int mt     = wave & 3;
    const int ntBase = (wave >> 2) * 8;

    const u16* wsW1  = wsb + WS_W1;
    const u16* wsW2  = wsb + WS_W2;
    const u16* wsHid = wsb + WS_HID;

    // ---- init: pad rows 49..63 of every A slice = 0, misc staging ----
    for (int idx = tid; idx < 16 * 240; idx += 256) {       // 240 u32 pad per slice
        int slice = idx / 240, rem = idx - slice * 240;
        ((u32*)sAfull)[slice * 1024 + 784 + rem] = 0u;      // 49*32/2 = 784
    }
    if (tid < H_) sAcc[tid] = 0.f;
    if (tid < T_ * MT) {
        int t = tid / MT, r = tid % MT;
        float w = 0.f;
        if (r < V_ - 1) {
            int s = (r < v) ? r : r + 1;
            int e = s * (V_ - 1) + v - (v > s ? 1 : 0);
            w = edges[((size_t)(b * E_ + e)) * ET_ + (t + 1)];
        }
        sEw[t][r] = w;
    }
    if (tid < D_) sX[tid] = inputs[(size_t)(b * V_ + v) * D_ + tid];

    // ---- stage ALL A slices once (async DMA): 16 slices x 49 rows x 4 chunks ----
    for (int idx = tid; idx < 16 * (V_ - 1) * 4; idx += 256) {
        int slice = idx / 196, rem = idx - slice * 196;
        int r = rem >> 2, q = rem & 3;
        int cg = slice * KS + q * 8;                  // global feature col
        int vs = (cg < H_) ? v : ((r < v) ? r : r + 1);
        int co = (cg < H_) ? cg : cg - H_;
        u32 voff = (u32)(((b * V_ + vs) * H_ + co) * 2);
        u32 ldsd = lds_off32(sAfull) + (u32)((slice * 2048 + r * KS + q * 8) * 2);
        async_cp16_s(ldsd, voff, wsHid);
    }

    // --- B staging: 16 KB block = 1024 chunks; 4 per thread via inst offsets ---
    auto stageB = [&](u16* buf, const u16* gblock) {
        u32 ldsd = lds_off32(buf) + (u32)tid * 16;
        u32 voff = (u32)tid * 16;
        asm volatile(
            "global_load_async_to_lds_b128 %0, %1, %2\n\t"
            "global_load_async_to_lds_b128 %0, %1, %2 offset:4096\n\t"
            "global_load_async_to_lds_b128 %0, %1, %2 offset:8192\n\t"
            "global_load_async_to_lds_b128 %0, %1, %2 offset:12288"
            :: "v"(ldsd), "v"(voff), "s"(gblock) : "memory");
    };

    v8f c[8];
    const int rowA = mt * 16 + (lane & 15);
    const int kbA  = (lane >> 4) * 16;

    // preload all 8 B fragments, then back-to-back WMMA chain
    auto compute8 = [&](const u16* pa, const u16* bufB) {
        Frag fa;
        fa.q[0] = *(const uint4*)pa;
        fa.q[1] = *(const uint4*)(pa + 8);
        Frag fb[8];
        #pragma unroll
        for (int i = 0; i < 8; ++i) {
            const uint4* p = (const uint4*)bufB + ((ntBase + i) * 32 + lane) * 2;
            fb[i].q[0] = p[0];
            fb[i].q[1] = p[1];
        }
        #pragma unroll
        for (int i = 0; i < 8; ++i)
            c[i] = __builtin_amdgcn_wmma_f32_16x16x32_bf16(
                       false, fa.v, false, fb[i].v, (short)0, c[i], false, false);
    };

    // first B block overlaps the A staging
    stageB(sB[0], wsW1);
    wait_async0();
    __syncthreads();

    for (int t = 0; t < T_; ++t) {
        // ======= GEMM1: m = tanh(A[64x512] @ W1[t]^T + b1[t]) =======
        #pragma unroll
        for (int i = 0; i < 8; ++i)
            #pragma unroll
            for (int g = 0; g < 8; ++g) c[i][g] = 0.f;

        for (int kk = 0; kk < 16; ++kk) {
            int cur = kk & 1, nxt = cur ^ 1;
            if (kk < 15)
                stageB(sB[nxt], wsW1 + (size_t)(t * 16 + kk + 1) * 8192);
            compute8(sAfull[kk] + rowA * KS + kbA, sB[cur]);
            wait_async0();
            __syncthreads();
        }
        // epilogue 1: tanh(c + b1) -> sM
        {
            int n_lo  = lane & 15;
            int m_off = (lane >> 4) * 8;
            #pragma unroll
            for (int i = 0; i < 8; ++i) {
                int h = (ntBase + i) * 16 + n_lo;
                float bias = b1[t * H_ + h];
                #pragma unroll
                for (int g = 0; g < 8; ++g) {
                    int row = mt * 16 + m_off + g;
                    sM[row * H_ + h] = f2bf(tanhf(c[i][g] + bias));
                }
            }
        }
        __syncthreads();

        // ======= GEMM2: c2 = sM[64x256] @ W2[t]^T =======
        #pragma unroll
        for (int i = 0; i < 8; ++i)
            #pragma unroll
            for (int g = 0; g < 8; ++g) c[i][g] = 0.f;

        stageB(sB[0], wsW2 + (size_t)(t * 8) * 8192);
        wait_async0();
        __syncthreads();
        for (int kk = 0; kk < 8; ++kk) {
            int cur = kk & 1, nxt = cur ^ 1;
            if (kk < 7)
                stageB(sB[nxt], wsW2 + (size_t)(t * 8 + kk + 1) * 8192);
            else if (t < T_ - 1)                          // prefetch next type's W1
                stageB(sB[nxt], wsW1 + (size_t)((t + 1) * 16) * 8192);
            compute8(sM + rowA * H_ + kk * KS + kbA, sB[cur]);
            wait_async0();
            __syncthreads();
        }
        // epilogue 2: tanh(c+b2)*ew, row-reduce, accumulate
        {
            int n_lo  = lane & 15;
            int m_off = (lane >> 4) * 8;
            #pragma unroll
            for (int i = 0; i < 8; ++i) {
                int h = (ntBase + i) * 16 + n_lo;
                float bias = b2[t * H_ + h];
                float part = 0.f;
                #pragma unroll
                for (int g = 0; g < 8; ++g) {
                    int row = mt * 16 + m_off + g;
                    part += tanhf(c[i][g] + bias) * sEw[t][row];
                }
                atomicAdd(&sAcc[h], part);
            }
        }
        __syncthreads();
        // note: sB[0] now holds next type's W1 block 0 (staged above)
        if (t < T_ - 1) {
            // re-stage the same block into sB[0], which already holds it; next
            // GEMM1 kk=0 reads sB[0], so this extra stage is redundant
            // (cheap, already resident in L2)
            stageB(sB[0], wsW1 + (size_t)((t + 1) * 16) * 8192);
            wait_async0();
            __syncthreads();
        }
    }

    // ======= GRU update + output MLP (one thread per channel) =======
    const int h = tid;
    const float invn = 1.f / ((float)T_ * (float)(V_ - 1));

    float dr = 0.f, di = 0.f, dn = 0.f;
    for (int k = 0; k < H_; ++k) {
        float a = sAcc[k] * invn;
        dr += Whr[h * H_ + k] * a;
        di += Whi[h * H_ + k] * a;
        dn += Whh[h * H_ + k] * a;
    }
    float xr = bir[h], xi = bii[h], xn = bin_w[h];
    #pragma unroll
    for (int k = 0; k < D_; ++k) {
        float xv = sX[k];
        xr += Wir[h * D_ + k] * xv;
        xi += Wii[h * D_ + k] * xv;
        xn += Win[h * D_ + k] * xv;
    }
    float rg = 1.f / (1.f + expf(-(xr + dr)));
    float ig = 1.f / (1.f + expf(-(xi + di)));
    float nn = tanhf(xn + rg * dn);
    float hprev = hidden[(size_t)(b * V_ + v) * H_ + h];
    float hn = (1.f - ig) * nn + ig * hprev;
    hid_out[(size_t)(b * V_ + v) * H_ + h] = hn;
    sVec[h] = hn;
    __syncthreads();

    float p1 = bo1[h];
    for (int k = 0; k < H_; ++k) p1 += Wo1[h * H_ + k] * sVec[k];
    p1 = fmaxf(p1, 0.f);
    sVec2[h] = p1;
    __syncthreads();

    float p2 = bo2[h];
    for (int k = 0; k < H_; ++k) p2 += Wo2[h * H_ + k] * sVec2[k];
    p2 = fmaxf(p2, 0.f);
    __syncthreads();
    sVec[h] = p2;
    __syncthreads();

    if (tid < D_) {
        float pv = bo3[tid];
        for (int k = 0; k < H_; ++k) pv += Wo3[tid * H_ + k] * sVec[k];
        pred_out[(size_t)(b * V_ + v) * D_ + tid] = sX[tid] + pv;
    }
}

extern "C" void kernel_launch(void* const* d_in, const int* in_sizes, int n_in,
                              void* d_out, int out_size, void* d_ws, size_t ws_size,
                              hipStream_t stream) {
    const float* inputs = (const float*)d_in[0];
    const float* hidden = (const float*)d_in[1];
    const float* edges  = (const float*)d_in[2];
    const float* W1     = (const float*)d_in[3];
    const float* b1     = (const float*)d_in[4];
    const float* W2     = (const float*)d_in[5];
    const float* b2     = (const float*)d_in[6];
    const float* Whr    = (const float*)d_in[7];
    const float* Whi    = (const float*)d_in[8];
    const float* Whh    = (const float*)d_in[9];
    const float* Wir    = (const float*)d_in[10];
    const float* bir    = (const float*)d_in[11];
    const float* Wii    = (const float*)d_in[12];
    const float* bii    = (const float*)d_in[13];
    const float* Win    = (const float*)d_in[14];
    const float* bin_w  = (const float*)d_in[15];
    const float* Wo1    = (const float*)d_in[16];
    const float* bo1    = (const float*)d_in[17];
    const float* Wo2    = (const float*)d_in[18];
    const float* bo2    = (const float*)d_in[19];
    const float* Wo3    = (const float*)d_in[20];
    const float* bo3    = (const float*)d_in[21];

    u16*   wsb  = (u16*)d_ws;
    float* pred = (float*)d_out;                        // [B,V,D]
    float* hidn = (float*)d_out + (size_t)B_ * V_ * D_; // [B,V,H]

    dnri_prep<<<(WS_TOT + 255) / 256, 256, 0, stream>>>(W1, W2, hidden, wsb);

    dnri_node_kernel<<<B_ * V_, 256, 0, stream>>>(
        inputs, hidden, edges, b1, b2, Whr, Whi, Whh,
        Wir, bir, Wii, bii, Win, bin_w, Wo1, bo1, Wo2, bo2, Wo3, bo3,
        wsb, pred, hidn);
}
